// HierarchicalSynthesisNet_60060822667662
// MI455X (gfx1250) — matrix operand order repaired and checked by reading, not compiled, not run on hardware
//
#include <hip/hip_runtime.h>
#include <math.h>

// ---------------------------------------------------------------------------
// CDNA5 (gfx1250, wave32) implementation of the hierarchical StyleGAN2-like
// synthesis network.  Modulated convs are folded to shared-weight GEMMs
// (input-scale by s, output-scale by demod d) executed with
// v_wmma_f32_16x16x32_bf16 through LDS-staged implicit-im2col tiles.
// K-order is permuted to kt*Cin+ic so all hot-loop index math is shift/mask.
// ---------------------------------------------------------------------------

typedef __bf16 bf16_t;
typedef __attribute__((ext_vector_type(16))) __bf16 v16bf;
typedef __attribute__((ext_vector_type(8)))  float  v8f;

#define FLAG_LEAKY 1
#define FLAG_ACC   2

// ------------------------------ small kernels ------------------------------

// permute W[oc][ic][kt] -> o[oc][kt*IC + ic], times scale, to bf16
__global__ void prep_weight_perm_kernel(const float* __restrict__ w,
                                        bf16_t* __restrict__ o, float scale,
                                        int OC, int IC, int ks2) {
  long n = (long)OC * IC * ks2;
  long i = (long)blockIdx.x * 256 + threadIdx.x;
  if (i >= n) return;
  long K = (long)IC * ks2;
  long oc = i / K;
  long rem = i - oc * K;
  long kt = rem / IC;
  long ic = rem - kt * IC;
  o[i] = (bf16_t)(w[(oc * IC + ic) * ks2 + kt] * scale);
}

// mean over HW per (b,c)
__global__ void pool_kernel(const float* __restrict__ z, float* __restrict__ o,
                            int n, int HW) {
  int i = blockIdx.x * 256 + threadIdx.x;
  if (i >= n) return;
  const float* p = z + (long)i * HW;
  float s = 0.f;
  for (int j = 0; j < HW; ++j) s += p[j];
  o[i] = s / (float)HW;
}

// out[b,o] = dot(in[b,:], w[o,:]) + bias[o]  (optionally leaky 0.2)
__global__ void linear_kernel(const float* __restrict__ in,
                              const float* __restrict__ w,
                              const float* __restrict__ bias,
                              float* __restrict__ o,
                              int B, int In, int Out, int leaky) {
  int i = blockIdx.x * 256 + threadIdx.x;
  if (i >= B * Out) return;
  int b = i / Out, oo = i - b * Out;
  const float* xi = in + (long)b * In;
  const float* wi = w + (long)oo * In;
  float acc = bias ? bias[oo] : 0.f;
  for (int j = 0; j < In; ++j) acc += xi[j] * wi[j];
  if (leaky) acc = acc > 0.f ? acc : 0.2f * acc;
  o[i] = acc;
}

// wsq[o,i] = sum_k (W[o,i,k]*scale)^2
__global__ void wsq_kernel(const float* __restrict__ w, float* __restrict__ o,
                           float scale, int OC, int IC, int ks2) {
  long i = (long)blockIdx.x * 256 + threadIdx.x;
  if (i >= (long)OC * IC) return;
  const float* p = w + i * ks2;
  float a = 0.f;
  for (int j = 0; j < ks2; ++j) { float t = p[j] * scale; a += t * t; }
  o[i] = a;
}

// d[b,o] = rsqrt(sum_i wsq[o,i]*s[b,i]^2 + 1e-8)
__global__ void dmod_kernel(const float* __restrict__ wsq,
                            const float* __restrict__ s, float* __restrict__ o,
                            int B, int OC, int IC) {
  int i = blockIdx.x * 256 + threadIdx.x;
  if (i >= B * OC) return;
  int b = i / OC, oc = i - b * OC;
  const float* sv = s + (long)b * IC;
  const float* wq = wsq + (long)oc * IC;
  float a = 1e-8f;
  for (int j = 0; j < IC; ++j) { float sj = sv[j]; a += wq[j] * sj * sj; }
  o[i] = rsqrtf(a);
}

// per-(b,c) instance norm, one 64-thread block per channel plane
__global__ void inorm_kernel(float* __restrict__ x, int HW) {
  __shared__ float red[64];
  long base = (long)blockIdx.x * HW;
  int t = threadIdx.x;
  float s = 0.f;
  for (int j = t; j < HW; j += 64) s += x[base + j];
  red[t] = s; __syncthreads();
  for (int k = 32; k > 0; k >>= 1) { if (t < k) red[t] += red[t + k]; __syncthreads(); }
  float mean = red[0] / (float)HW; __syncthreads();
  float v = 0.f;
  for (int j = t; j < HW; j += 64) { float d = x[base + j] - mean; v += d * d; }
  red[t] = v; __syncthreads();
  for (int k = 32; k > 0; k >>= 1) { if (t < k) red[t] += red[t + k]; __syncthreads(); }
  float inv = rsqrtf(red[0] / (float)HW + 1e-5f);
  __syncthreads();
  for (int j = t; j < HW; j += 64) x[base + j] = (x[base + j] - mean) * inv;
}

// bilinear 2x upsample, align_corners=False (half-pixel), edge clamp
__global__ void up2_kernel(const float* __restrict__ in, float* __restrict__ out,
                           int BC, int H, int W) {
  int OH = 2 * H, OW = 2 * W;
  long n = (long)BC * OH * OW;
  long i = (long)blockIdx.x * 256 + threadIdx.x;
  if (i >= n) return;
  int ow = (int)(i % OW); long t = i / OW;
  int oh = (int)(t % OH); int bc = (int)(t / OH);
  float sy = oh * 0.5f - 0.25f;
  float sx = ow * 0.5f - 0.25f;
  int y0 = (int)floorf(sy); float wy = sy - (float)y0;
  int x0 = (int)floorf(sx); float wx = sx - (float)x0;
  int y1 = y0 + 1, x1 = x0 + 1;
  y0 = max(0, min(H - 1, y0)); y1 = max(0, min(H - 1, y1));
  x0 = max(0, min(W - 1, x0)); x1 = max(0, min(W - 1, x1));
  const float* p = in + (long)bc * H * W;
  float v = (1.f - wy) * ((1.f - wx) * p[y0 * W + x0] + wx * p[y0 * W + x1]) +
            wy * ((1.f - wx) * p[y1 * W + x0] + wx * p[y1 * W + x1]);
  out[i] = v;
}

__global__ void tanh_kernel(const float* __restrict__ in, float* __restrict__ out, long n) {
  long i = (long)blockIdx.x * 256 + threadIdx.x;
  if (i < n) out[i] = tanhf(in[i]);
}

// --------------------------- WMMA conv GEMM kernel --------------------------
// Implicit-im2col GEMM:  out[m=(b,oh,ow), n=oc] = sum_k A[m,k]*Wt[n,k]
//   k = kt*Cin + ic,  kt = kh*ks+kw  (weights pre-permuted to this order)
//   A[m,k] = s[b,ic] * x[b,ic,oh+kh-p,ow+kw-p]
// Tile 256(M) x 64(N); 8 waves, each 64x32 via 4x2 wmma 16x16x32 bf16.
// All sizes are powers of two (except ks2<=9 handled by mul-shift), so the
// hot loop uses only shifts/masks for addressing.

#define TM 256
#define TN 64
#define TK 32

__global__ __launch_bounds__(256) void modconv_kernel(
    const float* __restrict__ x, const bf16_t* __restrict__ wt,
    const float* __restrict__ s, const float* __restrict__ dmod,
    const float* __restrict__ noise, const float* __restrict__ nscale,
    const float* __restrict__ bias, float* __restrict__ out,
    int Bn, int Cin, int Cout, int lgW, int lgCin, int ksz, int flags) {
  const int W = 1 << lgW;
  const int lgHW = 2 * lgW;
  const int HW = 1 << lgHW;
  const int M = Bn << lgHW;
  const int ks2 = ksz * ksz;
  const int K = Cin * ks2;
  const int pad = ksz >> 1;

  __shared__ bf16_t Asm[TM][TK + 2];
  __shared__ bf16_t Bsm[TK][TN + 2];

  const int tid = threadIdx.x;
  const int m0 = blockIdx.x * TM;
  const int n0 = blockIdx.y * TN;
  const int lane = tid & 31;
  const int wv = tid >> 5;
  const int wm = (wv >> 1) * 64;  // 0,64,128,192
  const int wn = (wv & 1) * 32;   // 0,32

  v8f acc[4][2];
#pragma unroll
  for (int i = 0; i < 4; ++i)
#pragma unroll
    for (int j = 0; j < 2; ++j)
#pragma unroll
      for (int r = 0; r < 8; ++r) acc[i][j][r] = 0.f;

  // fixed per-thread staging coordinates (256 threads, cols repeat exactly)
  const int ca = tid & 31;   // A column (k offset within TK)
  const int ra0 = tid >> 5;  // A row base, step 8 (32 passes)
  const int cb = tid & 63;   // B column (n offset within TN)
  const int rb0 = tid >> 6;  // B row base, step 4 (8 passes)
  const int nB = n0 + cb;
  const bool nok = nB < Cout;

  for (int k0 = 0; k0 < K; k0 += TK) {
    // ---- stage A tile (im2col + style scale + bf16 convert) ----
    {
      const int k = k0 + ca;              // k < K guaranteed (K % 32 == 0)
      const int ic = k & (Cin - 1);
      const int kt = k >> lgCin;
      const int kh = (kt * 11) >> 5;      // kt/3 for kt in [0,8]
      const int kw = kt - kh * 3;
      const long xic = (long)ic << lgHW;
#pragma unroll
      for (int i = 0; i < TM / 8; ++i) {
        int r = ra0 + i * 8;
        int m = m0 + r;                   // m < M guaranteed (M % 256 == 0)
        int b = m >> lgHW;
        int hw = m & (HW - 1);
        int oh = hw >> lgW;
        int ow = hw & (W - 1);
        int ih = oh + kh - pad, iw = ow + kw - pad;
        float v = 0.f;
        if (ih >= 0 && ih < W && iw >= 0 && iw < W) {
          v = x[(((long)b * Cin) << lgHW) + xic + (ih << lgW) + iw];
          if (s) v *= s[(b << lgCin) + ic];
        }
        Asm[r][ca] = (bf16_t)v;
      }
    }
    // ---- stage B tile (pre-permuted bf16 weights, [Cout][K] row-major) ----
#pragma unroll
    for (int i = 0; i < TK / 4; ++i) {
      int r = rb0 + i * 4;
      bf16_t v = (bf16_t)0.f;
      if (nok) v = wt[(long)nB * K + k0 + r];
      Bsm[r][cb] = v;
    }
    __syncthreads();

    // ---- fragments in contiguous k order (k = 16*half + j) for A and B, then WMMA ----
    const int arow = lane & 15;
    const int khiA = (lane >> 4) * 16;
    const int bcol = lane & 15;
    const int khiB = (lane >> 4) * 16;
    v16bf afrag[4], bfrag[2];
#pragma unroll
    for (int im = 0; im < 4; ++im) {
      int rb = wm + im * 16 + arow;
#pragma unroll
      for (int j = 0; j < 16; ++j) afrag[im][j] = Asm[rb][khiA + j];
    }
#pragma unroll
    for (int in_ = 0; in_ < 2; ++in_) {
      int cc = wn + in_ * 16 + bcol;
#pragma unroll
      for (int j = 0; j < 16; ++j) bfrag[in_][j] = Bsm[khiB + j][cc];
    }
#pragma unroll
    for (int im = 0; im < 4; ++im)
#pragma unroll
      for (int in_ = 0; in_ < 2; ++in_)
        acc[im][in_] = __builtin_amdgcn_wmma_f32_16x16x32_bf16(
            false, afrag[im], false, bfrag[in_], (short)0, acc[im][in_],
            false, false);
    __syncthreads();
  }

  // ---- epilogue: demod * acc + bias + noise*ns, leaky, store/accumulate ----
  const bool leaky = (flags & FLAG_LEAKY) != 0;
  const bool accum = (flags & FLAG_ACC) != 0;
  const float nsv = nscale ? nscale[0] : 0.f;
#pragma unroll
  for (int im = 0; im < 4; ++im)
#pragma unroll
    for (int in_ = 0; in_ < 2; ++in_)
#pragma unroll
      for (int r = 0; r < 8; ++r) {
        int m = m0 + wm + im * 16 + (lane >> 4) * 8 + r;
        int n = n0 + wn + in_ * 16 + (lane & 15);
        if (n >= Cout) continue;
        int b = m >> lgHW;
        int hw = m & (HW - 1);
        float v = acc[im][in_][r];
        if (dmod) v *= dmod[b * Cout + n];
        if (bias) v += bias[n];
        if (noise) v += noise[((long)b << lgHW) + hw] * nsv;
        if (leaky) v = v > 0.f ? v : 0.2f * v;
        long o = ((long)(b * Cout + n) << lgHW) + hw;
        if (accum) out[o] += v; else out[o] = v;
      }
}

// ------------------------------- host driver -------------------------------

static inline unsigned nb(long n) { return (unsigned)((n + 255) / 256); }
static inline int ilog2i(int v) { int l = 0; while ((1 << l) < v) ++l; return l; }

extern "C" void kernel_launch(void* const* d_in, const int* in_sizes, int n_in,
                              void* d_out, int out_size, void* d_ws, size_t ws_size,
                              hipStream_t stream) {
  (void)in_sizes; (void)n_in; (void)out_size; (void)ws_size;
  const int B = 32;

  // ---- input table (setup_inputs() dict insertion order, leaves in order) ----
  const float* z_t = (const float*)d_in[0];
  const float* z_m = (const float*)d_in[1];
  const float* z_b = (const float*)d_in[2];
  const float* map_w[6]; const float* map_b[6];
  for (int l = 0; l < 6; ++l) { map_w[l] = (const float*)d_in[3 + 2 * l];
                                map_b[l] = (const float*)d_in[4 + 2 * l]; }
  const float* proj_w = (const float*)d_in[15];
  const float* proj_b = (const float*)d_in[16];
  const float* injm_w = (const float*)d_in[17];
  const float* injm_b = (const float*)d_in[18];
  const float* injb_w = (const float*)d_in[19];
  const float* injb_b = (const float*)d_in[20];

  struct ModL { const float *w, *mw, *mb, *ns; int oc, ic, ks; };
  auto getL = [&](int base, int oc, int ic, int ks) {
    ModL L; L.w = (const float*)d_in[base]; L.mw = (const float*)d_in[base + 1];
    L.mb = (const float*)d_in[base + 2]; L.ns = (const float*)d_in[base + 3];
    L.oc = oc; L.ic = ic; L.ks = ks; return L;
  };
  ModL c8   = getL(21, 512, 512, 3), r8   = getL(25, 3, 512, 1);
  ModL c16  = getL(29, 512, 512, 3), r16  = getL(33, 3, 512, 1);
  ModL c32  = getL(37, 256, 512, 3), r32  = getL(41, 3, 256, 1);
  ModL c64  = getL(45, 128, 256, 3), r64  = getL(49, 3, 128, 1);
  ModL c128 = getL(53,  64, 128, 3), r128 = getL(57, 3,  64, 1);
  const float* noi[10]; for (int i = 0; i < 10; ++i) noi[i] = (const float*)d_in[61 + i];
  // noi: 0=c8 1=r8 2=c16 3=r16 4=c32 5=r32 6=c64 7=r64 8=c128 9=r128

  // ---- workspace carve ----
  char* wsb = (char*)d_ws; size_t off = 0;
  auto carve = [&](size_t bytes) -> void* {
    off = (off + 255) & ~(size_t)255; void* p = wsb + off; off += bytes; return p;
  };
  float* bufA  = (float*)carve((size_t)67108864 * 4);   // up to 32x128x128x128
  float* bufB  = (float*)carve((size_t)33554432 * 4);   // up to 32x64x128x128
  float* rgbA  = (float*)carve((size_t)1572864 * 4);
  float* rgbB  = (float*)carve((size_t)1572864 * 4);
  float* wmapA = (float*)carve(16384 * 4);
  float* wmapB = (float*)carve(16384 * 4);
  float* sbuf  = (float*)carve(16384 * 4);
  float* dbuf  = (float*)carve(16384 * 4);
  float* wsqb  = (float*)carve(262144 * 4);
  auto carve_bf = [&](long n) { return (bf16_t*)carve((size_t)n * 2); };
  bf16_t* wp_proj = carve_bf(512 * 64);
  bf16_t* wp_injm = carve_bf(512 * 64);
  bf16_t* wp_injb = carve_bf(512 * 64);
  bf16_t* wp_c8   = carve_bf((long)512 * 4608);
  bf16_t* wp_r8   = carve_bf(3 * 512);
  bf16_t* wp_c16  = carve_bf((long)512 * 4608);
  bf16_t* wp_r16  = carve_bf(3 * 512);
  bf16_t* wp_c32  = carve_bf((long)256 * 4608);
  bf16_t* wp_r32  = carve_bf(3 * 256);
  bf16_t* wp_c64  = carve_bf((long)128 * 2304);
  bf16_t* wp_r64  = carve_bf(3 * 128);
  bf16_t* wp_c128 = carve_bf((long)64 * 1152);
  bf16_t* wp_r128 = carve_bf(3 * 64);

  auto prep = [&](const float* w, bf16_t* o, int oc, int ic, int ks, float sc) {
    long n = (long)oc * ic * ks * ks;
    prep_weight_perm_kernel<<<nb(n), 256, 0, stream>>>(w, o, sc, oc, ic, ks * ks);
  };
  auto scl = [](int ic, int ks) { return 1.f / sqrtf((float)(ic * ks * ks)); };
  prep(proj_w, wp_proj, 512, 64, 1, 1.f);
  prep(injm_w, wp_injm, 512, 64, 1, 1.f);
  prep(injb_w, wp_injb, 512, 64, 1, 1.f);
  prep(c8.w,   wp_c8,   512, 512, 3, scl(512, 3)); prep(r8.w,   wp_r8,   3, 512, 1, scl(512, 1));
  prep(c16.w,  wp_c16,  512, 512, 3, scl(512, 3)); prep(r16.w,  wp_r16,  3, 512, 1, scl(512, 1));
  prep(c32.w,  wp_c32,  256, 512, 3, scl(512, 3)); prep(r32.w,  wp_r32,  3, 256, 1, scl(256, 1));
  prep(c64.w,  wp_c64,  128, 256, 3, scl(256, 3)); prep(r64.w,  wp_r64,  3, 128, 1, scl(128, 1));
  prep(c128.w, wp_c128,  64, 128, 3, scl(128, 3)); prep(r128.w, wp_r128, 3,  64, 1, scl(64, 1));

  // ---- mapping network (fp32 exact) ----
  pool_kernel<<<nb(B * 64), 256, 0, stream>>>(z_t, wmapA, B * 64, 64);
  linear_kernel<<<nb(B * 512), 256, 0, stream>>>(wmapA, map_w[0], map_b[0], wmapB, B, 64, 512, 1);
  float* cur = wmapB; float* nxt = wmapA;
  for (int l = 1; l < 6; ++l) {
    linear_kernel<<<nb(B * 512), 256, 0, stream>>>(cur, map_w[l], map_b[l], nxt, B, 512, 512, 1);
    float* t = cur; cur = nxt; nxt = t;
  }
  const float* wstyle = cur;  // [32,512]

  // ---- generic launchers ----
  auto conv = [&](const float* x, const bf16_t* wt, const float* sv, const float* dv,
                  const float* nz, const float* nsp, const float* bias, float* out,
                  int Cin, int Cout, int H, int ks, int flags) {
    int M = B * H * H;
    dim3 g((unsigned)((M + TM - 1) / TM), (unsigned)((Cout + TN - 1) / TN));
    modconv_kernel<<<g, 256, 0, stream>>>(x, wt, sv, dv, nz, nsp, bias, out,
                                          B, Cin, Cout, ilog2i(H), ilog2i(Cin), ks, flags);
  };
  auto mod_stage = [&](const ModL& L, const bf16_t* wbf, const float* x,
                       const float* nz, float* out, int H, bool demod, int extra) {
    linear_kernel<<<nb(B * L.ic), 256, 0, stream>>>(wstyle, L.mw, L.mb, sbuf, B, 512, L.ic, 0);
    const float* dp = nullptr;
    if (demod) {
      wsq_kernel<<<nb((long)L.oc * L.ic), 256, 0, stream>>>(L.w, wsqb, scl(L.ic, L.ks),
                                                            L.oc, L.ic, L.ks * L.ks);
      dmod_kernel<<<nb(B * L.oc), 256, 0, stream>>>(wsqb, sbuf, dbuf, B, L.oc, L.ic);
      dp = dbuf;
    }
    conv(x, wbf, sbuf, dp, nz, L.ns, nullptr, out, L.ic, L.oc, H, L.ks, FLAG_LEAKY | extra);
  };
  auto up = [&](const float* in, float* out, int C, int H) {
    long n = (long)B * C * 4 * H * H;
    up2_kernel<<<nb(n), 256, 0, stream>>>(in, out, B * C, H, H);
  };

  // ---- synthesis ----
  conv(z_t, wp_proj, nullptr, nullptr, nullptr, nullptr, proj_b, bufA, 64, 512, 8, 1, FLAG_LEAKY);
  inorm_kernel<<<B * 512, 64, 0, stream>>>(bufA, 64);

  mod_stage(c8, wp_c8, bufA, noi[0], bufB, 8, true, 0);          // conv8 -> bufB
  mod_stage(r8, wp_r8, bufB, noi[1], rgbA, 8, false, 0);         // rgb8  -> rgbA

  up(bufB, bufA, 512, 8);  up(rgbA, rgbB, 3, 8);                 // -> 16
  conv(z_m, wp_injm, nullptr, nullptr, nullptr, nullptr, injm_b, bufA, 64, 512, 16, 1, FLAG_ACC);
  mod_stage(c16, wp_c16, bufA, noi[2], bufB, 16, true, 0);
  mod_stage(r16, wp_r16, bufB, noi[3], rgbB, 16, false, FLAG_ACC);

  up(bufB, bufA, 512, 16); up(rgbB, rgbA, 3, 16);                // -> 32
  conv(z_b, wp_injb, nullptr, nullptr, nullptr, nullptr, injb_b, bufA, 64, 512, 32, 1, FLAG_ACC);
  mod_stage(c32, wp_c32, bufA, noi[4], bufB, 32, true, 0);       // 256ch
  mod_stage(r32, wp_r32, bufB, noi[5], rgbA, 32, false, FLAG_ACC);

  up(bufB, bufA, 256, 32); up(rgbA, rgbB, 3, 32);                // -> 64
  mod_stage(c64, wp_c64, bufA, noi[6], bufB, 64, true, 0);       // 128ch
  mod_stage(r64, wp_r64, bufB, noi[7], rgbB, 64, false, FLAG_ACC);

  up(bufB, bufA, 128, 64); up(rgbB, rgbA, 3, 64);                // -> 128
  mod_stage(c128, wp_c128, bufA, noi[8], bufB, 128, true, 0);    // 64ch
  mod_stage(r128, wp_r128, bufB, noi[9], rgbA, 128, false, FLAG_ACC);

  long nout = (long)B * 3 * 128 * 128;
  tanh_kernel<<<nb(nout), 256, 0, stream>>>(rgbA, (float*)d_out, nout);
}
